// WLS_net_v1_89515708383699
// MI455X (gfx1250) — matrix operand order repaired and checked by reading, not compiled, not run on hardware
//
#include <hip/hip_runtime.h>

typedef _Float16 h16;
typedef __attribute__((ext_vector_type(16))) _Float16 v16h;
typedef __attribute__((ext_vector_type(8)))  _Float16 v8h;
typedef __attribute__((ext_vector_type(8)))  float    v8f;
typedef __attribute__((ext_vector_type(4)))  unsigned int u32x4;
typedef __attribute__((ext_vector_type(8)))  int      i32x8;
typedef __attribute__((ext_vector_type(4)))  int      i32x4;

#define HH 128
#define WW 128
#define PW 130            // padded spatial extent (1-px zero halo ring)
#define CPAD 128          // padded input channels (K-dim), multiple of 32
#define NCT 4             // co 16-tiles per block group (co padded to groups of 64)
#define NPX 4             // pixel 16-tiles per wave (A-fragment reuse factor)
#define WOUT 448          // padded 441 softmax channels

// ---------------------------------------------------------------------------
// Pack NCHW f32 input (10 ch) into zero-halo channel-last f16 [130][130][128]
// ---------------------------------------------------------------------------
__global__ __launch_bounds__(256) void pack_input_kernel(const float* __restrict__ in,
                                                         h16* __restrict__ act) {
  int idx = blockIdx.x * 256 + threadIdx.x;   // (yp*130+xp)*128 + ci
  int ci = idx & (CPAD - 1);
  int xy = idx >> 7;
  int xp = xy % PW;
  int yp = xy / PW;
  float v = 0.f;
  if (ci < 10 && xp >= 1 && xp <= WW && yp >= 1 && yp <= HH)
    v = in[ci * (HH * WW) + (yp - 1) * WW + (xp - 1)];
  act[idx] = (h16)v;
}

// ---------------------------------------------------------------------------
// Repack OIHW f32 weights into WMMA A-fragment order:
//   wpf[ct][tap=9][kc][lane=32][i=16 halves], per-lane contiguous K (k = 16*(lane/16) + i) baked in.
// ---------------------------------------------------------------------------
__global__ __launch_bounds__(256) void pack_weights_kernel(const float* __restrict__ w,
                                                           const float* __restrict__ b,
                                                           h16* __restrict__ wpf,
                                                           float* __restrict__ bp,
                                                           int CO, int CI,
                                                           int ntiles, int nkc) {
  int idx = blockIdx.x * 256 + threadIdx.x;
  int total = ntiles * 9 * nkc * 512;
  if (idx < total) {
    int i = idx & 15;
    int lane = (idx >> 4) & 31;
    int frag = idx >> 9;
    int kc = frag % nkc;
    int t = (frag / nkc) % 9;            // kh*3 + kw
    int ct = frag / (nkc * 9);
    int m = lane & 15;
    int co = ct * 16 + m;
    int kb = (lane < 16) ? 0 : 16;
    int koff = kb + i;
    int ci = kc * 32 + koff;
    float v = (co < CO && ci < CI) ? w[((co * CI + ci) * 9) + t] : 0.f;
    wpf[idx] = (h16)v;
  }
  if (idx < ntiles * 16) bp[idx] = (idx < CO) ? b[idx] : 0.f;
}

// ---------------------------------------------------------------------------
// Per-tile epilogue store
// ---------------------------------------------------------------------------
__device__ __forceinline__ void conv_store_tile(const v8f& a, const float* __restrict__ bp,
                                                h16* __restrict__ actOutH,
                                                float* __restrict__ outF,
                                                int y, int x, int cobase, int mode) {
  if (mode == 2) {
    float tmp[8];
#pragma unroll
    for (int r = 0; r < 8; ++r) tmp[r] = a[r] + bp[cobase + r];
    float* o = outF + ((size_t)(y * WW + x) * WOUT + cobase);
    *(float4*)(o) = make_float4(tmp[0], tmp[1], tmp[2], tmp[3]);
    *(float4*)(o + 4) = make_float4(tmp[4], tmp[5], tmp[6], tmp[7]);
  } else {
    v8h hv;
#pragma unroll
    for (int r = 0; r < 8; ++r) {
      float v = a[r] + bp[cobase + r];
      if (mode == 1) v = (v > 0.f) ? v : 0.01f * v;
      hv[r] = (h16)v;
    }
    *(v8h*)(actOutH + (((y + 1) * PW + (x + 1)) * CPAD + cobase)) = hv;
  }
}

// ---------------------------------------------------------------------------
// 3x3 same-padding conv as implicit GEMM with WMMA f16 -> f32 accum.
//   Weight slice (NCT*9*nkc KB, <=147 KB) staged into LDS by the Tensor Data
//   Mover (wave 0 issues TENSOR_LOAD_TO_LDS; TENSORcnt wait; block barrier).
//   Each A fragment (ds_load_b128 x2) is reused by NPX=4 pixel tiles ->
//   256 B/wmma LDS traffic. Each B fragment (global 32 B/lane) is reused by
//   NCT=4 wmmas -> 256 B/wmma from L2.
//   mode 0: f16 out, no act   mode 1: f16 out + leaky(0.01)   mode 2: f32 out
// Grid: (co groups, HH/2); block 128 = 4 waves.
//   wave w -> row = 2*blockIdx.y + (w>>1), pixels (w&1)*64 .. +63 (four tiles)
// ---------------------------------------------------------------------------
__global__ __launch_bounds__(128) void conv3x3_wmma_kernel(const h16* __restrict__ actIn,
                                                           const h16* __restrict__ wpf,
                                                           const float* __restrict__ bp,
                                                           h16* __restrict__ actOutH,
                                                           float* __restrict__ outF,
                                                           int nkc, int mode) {
  extern __shared__ char smem[];
  h16* wlds = (h16*)smem;                       // [NCT][9][nkc][32][16] halves
  const int tid = threadIdx.x;
  const int ct_base = blockIdx.x * NCT;
  const h16* gsrc = wpf + (size_t)blockIdx.x * NCT * 9 * nkc * 512;

#if __has_builtin(__builtin_amdgcn_tensor_load_to_lds)
  // Tensor Data Mover: DMA the weight slice into LDS. 1-row 2D descriptor,
  // 8-byte elements; tile_dim0 = n8 (<= 65535 for nkc<=4). Issued by wave 0
  // (EXEC ignored by tensor ops); completion via TENSORcnt, then barrier.
  if (tid < 32) {
    const int n8 = NCT * 9 * nkc * 128;         // 8-byte elements (1 KB/frag)
    unsigned long long ga = (unsigned long long)(uintptr_t)gsrc;
    u32x4 g0;
    g0[0] = 0x1u;                                // count=1, user mode
    g0[1] = 0u;                                  // lds_addr: dynamic-LDS base
    g0[2] = (unsigned)(ga & 0xffffffffu);        // global_addr[31:0]
    g0[3] = (unsigned)((ga >> 32) & 0x1ffffffu) | (2u << 30);  // addr hi | type=2
    i32x8 g1;
    g1[0] = 0x00030000;                          // data_size=8B, mask=0
    g1[1] = (n8 & 0xffff) << 16;                 // tensor_dim0 lo16
    g1[2] = (n8 >> 16) | (1 << 16);              // tensor_dim0 hi | tensor_dim1=1 lo
    g1[3] = (n8 & 0xffff) << 16;                 // tensor_dim1 hi=0 | tile_dim0
    g1[4] = 1;                                   // tile_dim1=1, tile_dim2=0
    g1[5] = n8;                                  // tensor_dim0_stride lo32
    g1[6] = 0;                                   // stride hi | dim1_stride lo
    g1[7] = 0;
    i32x4 gz4 = {0, 0, 0, 0};                    // groups 2/3: unused (2D tensor)
    i32x8 gz8 = {0, 0, 0, 0, 0, 0, 0, 0};        // extra descriptor group (unused)
    __builtin_amdgcn_tensor_load_to_lds(g0, g1, gz4, gz4, gz8, 0);
    __builtin_amdgcn_s_wait_tensorcnt(0);
  }
#else
  {
    const int n16 = NCT * 9 * nkc * 64;          // uint4 count (512 halves/frag)
    const uint4* src = (const uint4*)gsrc;
    uint4* dst = (uint4*)wlds;
    for (int i = tid; i < n16; i += 128) dst[i] = src[i];
  }
#endif
  __syncthreads();

  const int wave = tid >> 5;
  const int lane = tid & 31;
  const int y = blockIdx.y * 2 + (wave >> 1);   // image row for this wave
  const int px0 = (wave & 1) * 64;              // four 16-px tiles: px0+16k
  const int nl = lane & 15;
  const int kbB = (lane < 16) ? 0 : 16;         // B fragment K split 0..15 / 16..31

  v8f zf = {0.f, 0.f, 0.f, 0.f, 0.f, 0.f, 0.f, 0.f};
  v8f acc[NCT][NPX];
#pragma unroll
  for (int c = 0; c < NCT; ++c)
#pragma unroll
    for (int k = 0; k < NPX; ++k) acc[c][k] = zf;

#pragma unroll
  for (int t = 0; t < 9; ++t) {
    const int dyt = t / 3, dxt = t % 3;
    // padded coords: input row = y + dyt (0..129), col = px0+nl+dxt (0..129)
    const h16* brow = actIn + (((y + dyt) * PW + (px0 + nl + dxt)) * CPAD);
    for (int kc = 0; kc < nkc; ++kc) {
      v16h B[NPX];
#pragma unroll
      for (int k = 0; k < NPX; ++k)
        B[k] = *(const v16h*)(brow + k * 16 * CPAD + kc * 32 + kbB);
#pragma unroll
      for (int c = 0; c < NCT; ++c) {
        const int frag = (c * 9 + t) * nkc + kc;
        v16h A = *(const v16h*)(wlds + frag * 512 + lane * 16);
#pragma unroll
        for (int k = 0; k < NPX; ++k)
          acc[c][k] = __builtin_amdgcn_wmma_f32_16x16x32_f16(
              false, A, false, B[k], (short)0, acc[c][k], false, false);
      }
    }
  }

  // D: lane -> col N = pixel; VGPR r -> row M = kbD + r (= output channel)
  const int kbD = (lane < 16) ? 0 : 8;
#pragma unroll
  for (int c = 0; c < NCT; ++c) {
    const int cobase = (ct_base + c) * 16 + kbD;
#pragma unroll
    for (int k = 0; k < NPX; ++k)
      conv_store_tile(acc[c][k], bp, actOutH, outF, y, px0 + k * 16 + nl, cobase, mode);
  }
}

// ---------------------------------------------------------------------------
// Softmax over 441 channels, channel-last [pix][448], one wave32 per pixel
// ---------------------------------------------------------------------------
__global__ __launch_bounds__(256) void softmax441_kernel(float* __restrict__ W) {
  int pix = blockIdx.x * 8 + (threadIdx.x >> 5);
  int lane = threadIdx.x & 31;
  float* p = W + (size_t)pix * WOUT;
  float m = -1e30f;
  for (int c = lane; c < 441; c += 32) m = fmaxf(m, p[c]);
#pragma unroll
  for (int o = 16; o > 0; o >>= 1) m = fmaxf(m, __shfl_xor(m, o, 32));
  float s = 0.f;
  for (int c = lane; c < 441; c += 32) {
    float e = __expf(p[c] - m);
    p[c] = e;
    s += e;
  }
#pragma unroll
  for (int o = 16; o > 0; o >>= 1) s += __shfl_xor(s, o, 32);
  float inv = 1.f / s;
  for (int c = lane; c < 441; c += 32) p[c] *= inv;
}

// ---------------------------------------------------------------------------
// Per-pixel WLS: A = sum_k w_k d_k d_k^T (+eps reg), B = sum_k w_k d_k y_k^T,
// solve 8x8 (3 rhs) by Gaussian elimination, predict at window center.
// ---------------------------------------------------------------------------
__global__ __launch_bounds__(256) void wls_solve_kernel(const float* __restrict__ in,
                                                        const float* __restrict__ W,
                                                        float* __restrict__ out) {
  int pix = blockIdx.x * blockDim.x + threadIdx.x;   // 0..16383
  int y = pix >> 7, x = pix & 127;
  const float* wrow = W + (size_t)pix * WOUT;

  float A[8][8];
  float Bm[8][3];
#pragma unroll
  for (int i = 0; i < 8; ++i) {
#pragma unroll
    for (int j = 0; j < 8; ++j) A[i][j] = 0.f;
#pragma unroll
    for (int c = 0; c < 3; ++c) Bm[i][c] = 0.f;
  }

  for (int dy = -10; dy <= 10; ++dy) {
    int yy = y + dy;
    if ((unsigned)yy >= (unsigned)HH) continue;   // zero-padded window: d==0, skip
    for (int dx = -10; dx <= 10; ++dx) {
      int xx = x + dx;
      if ((unsigned)xx >= (unsigned)WW) continue;
      float w = wrow[(dy + 10) * 21 + (dx + 10)];
      int off = yy * WW + xx;
      float d[8];
      d[0] = 1.f;
#pragma unroll
      for (int j = 1; j < 8; ++j) d[j] = in[(2 + j) * (HH * WW) + off];  // input ch 3..9
      float yv[3];
#pragma unroll
      for (int c = 0; c < 3; ++c) yv[c] = in[c * (HH * WW) + off];
      float wd[8];
#pragma unroll
      for (int i = 0; i < 8; ++i) wd[i] = w * d[i];
#pragma unroll
      for (int i = 0; i < 8; ++i) {
#pragma unroll
        for (int j = 0; j < 8; ++j) A[i][j] += wd[i] * d[j];
#pragma unroll
        for (int c = 0; c < 3; ++c) Bm[i][c] += wd[i] * yv[c];
      }
    }
  }
#pragma unroll
  for (int i = 1; i < 8; ++i) A[i][i] += 1e-4f;   // reg, [0][0] excluded

  // Gaussian elimination (no pivot; A[0][0] = sum of softmax weights > 0)
#pragma unroll
  for (int k = 0; k < 8; ++k) {
    float inv = 1.f / A[k][k];
#pragma unroll
    for (int j = k + 1; j < 8; ++j) A[k][j] *= inv;
#pragma unroll
    for (int c = 0; c < 3; ++c) Bm[k][c] *= inv;
#pragma unroll
    for (int i = k + 1; i < 8; ++i) {
      float f = A[i][k];
#pragma unroll
      for (int j = k + 1; j < 8; ++j) A[i][j] -= f * A[k][j];
#pragma unroll
      for (int c = 0; c < 3; ++c) Bm[i][c] -= f * Bm[k][c];
    }
  }
#pragma unroll
  for (int i = 6; i >= 0; --i) {
#pragma unroll
    for (int c = 0; c < 3; ++c) {
      float v = Bm[i][c];
#pragma unroll
      for (int j = i + 1; j < 8; ++j) v -= A[i][j] * Bm[j][c];
      Bm[i][c] = v;
    }
  }

  // prediction with center design vector
  float d0[8];
  d0[0] = 1.f;
#pragma unroll
  for (int j = 1; j < 8; ++j) d0[j] = in[(2 + j) * (HH * WW) + pix];
#pragma unroll
  for (int c = 0; c < 3; ++c) {
    float v = 0.f;
#pragma unroll
    for (int i = 0; i < 8; ++i) v += d0[i] * Bm[i][c];
    out[c * (HH * WW) + pix] = v;
  }
}

// ---------------------------------------------------------------------------
extern "C" void kernel_launch(void* const* d_in, const int* in_sizes, int n_in,
                              void* d_out, int out_size, void* d_ws, size_t ws_size,
                              hipStream_t stream) {
  (void)in_sizes; (void)n_in; (void)out_size; (void)ws_size;
  const float* input = (const float*)d_in[0];   // [1,10,128,128]
  // d_in[1] = ref (unused by only_img_out path)
  const float* w0    = (const float*)d_in[2];   // [100,10,3,3]
  const float* b0    = (const float*)d_in[3];   // [100]
  const float* wmid  = (const float*)d_in[4];   // [23,100,100,3,3]
  const float* bmid  = (const float*)d_in[5];   // [23,100]
  const float* wfin  = (const float*)d_in[6];   // [441,100,3,3]
  const float* bfin  = (const float*)d_in[7];   // [441]
  float* out = (float*)d_out;                   // [1,3,128,128]

  char* ws = (char*)d_ws;
  const size_t ACT_BYTES = (size_t)PW * PW * CPAD * sizeof(h16);   // 4.33 MB padded
  h16* actA   = (h16*)(ws);
  h16* actB   = (h16*)(ws + ACT_BYTES);
  h16* wpf    = (h16*)(ws + 2 * ACT_BYTES);                        // <= 1.0 MB used
  float* bp   = (float*)(ws + 2 * ACT_BYTES + (2u << 20));
  float* Wbuf = (float*)(ws + 2 * ACT_BYTES + (2u << 20) + 65536); // 16384*448*4 = 28 MB

  const int SH4 = NCT * 9 * 4 * 512 * (int)sizeof(h16);            // 147,456 B (nkc=4)
  const int SH1 = NCT * 9 * 1 * 512 * (int)sizeof(h16);            // 36,864 B  (nkc=1)
  (void)hipFuncSetAttribute((const void*)conv3x3_wmma_kernel,
                            hipFuncAttributeMaxDynamicSharedMemorySize, SH4);

  // actB: zero halo ring (interior incl. pad channels is fully written by convs)
  (void)hipMemsetAsync(actB, 0, ACT_BYTES, stream);
  pack_input_kernel<<<((int)(PW * PW * CPAD)) / 256, 256, 0, stream>>>(input, actA);

  // conv0: 10 -> 100 (co padded to 128 = 2 groups of NCT), single K chunk
  pack_weights_kernel<<<(8 * 9 * 1 * 512 + 255) / 256, 256, 0, stream>>>(
      w0, b0, wpf, bp, 100, 10, 8, 1);
  conv3x3_wmma_kernel<<<dim3(2, HH / 2), 128, SH1, stream>>>(actA, wpf, bp, actB, nullptr, 1, 0);

  // 23 mid layers: 100 -> 100 (co padded to 128), leaky relu
  h16* cur = actB; h16* nxt = actA;
  for (int l = 0; l < 23; ++l) {
    pack_weights_kernel<<<(8 * 9 * 4 * 512 + 255) / 256, 256, 0, stream>>>(
        wmid + (size_t)l * 100 * 100 * 9, bmid + l * 100, wpf, bp, 100, 100, 8, 4);
    conv3x3_wmma_kernel<<<dim3(2, HH / 2), 128, SH4, stream>>>(cur, wpf, bp, nxt, nullptr, 4, 1);
    h16* t = cur; cur = nxt; nxt = t;
  }

  // final conv: 100 -> 441 (28 co-tiles in 7 groups of NCT), f32 output
  pack_weights_kernel<<<(28 * 9 * 4 * 512 + 255) / 256, 256, 0, stream>>>(
      wfin, bfin, wpf, bp, 441, 100, 28, 4);
  conv3x3_wmma_kernel<<<dim3(7, HH / 2), 128, SH4, stream>>>(cur, wpf, bp, nullptr, Wbuf, 4, 2);

  softmax441_kernel<<<(HH * WW) / 8, 256, 0, stream>>>(Wbuf);
  wls_solve_kernel<<<(HH * WW) / 256, 256, 0, stream>>>(input, Wbuf, out);
}
